// GIN_encoder_10969346474303
// MI455X (gfx1250) — matrix operand order repaired and checked by reading, not compiled, not run on hardware
//
#include <hip/hip_runtime.h>
#include <hip/hip_bf16.h>

typedef __attribute__((ext_vector_type(16))) __bf16 v16bf;
typedef __attribute__((ext_vector_type(8)))  float  v8f;

#define BN_EPS 1e-5f

constexpr int Fdim = 512;            // input features (K of GEMM)
constexpr int Hdim = 512;            // hidden features (N of GEMM)
constexpr int KCHUNKS = Fdim / 32;   // 16 K-chunks of 32 for bf16 WMMA

union BfBits { __bf16 b; unsigned short s; };

__device__ inline unsigned pack_hilo(float v) {
  BfBits hi, lo;
  hi.b = (__bf16)v;
  lo.b = (__bf16)(v - (float)hi.b);
  return (unsigned)hi.s | ((unsigned)lo.s << 16);
}

// ---------------- kernel 1: zero BN stat accumulators ----------------
__global__ void zero_stats_kernel(float* __restrict__ stats, int n) {
  int i = blockIdx.x * blockDim.x + threadIdx.x;
  if (i < n) stats[i] = 0.0f;
}

// ---------------- kernel 2: split W into bf16 hi/lo planes (one-time, 1 MB) ----------------
__global__ void wsplit_kernel(const float* __restrict__ W,
                              __bf16* __restrict__ whi, __bf16* __restrict__ wlo,
                              int n) {
  int i = blockIdx.x * blockDim.x + threadIdx.x;
  if (i >= n) return;
  float v = W[i];
  __bf16 hi = (__bf16)v;
  whi[i] = hi;
  wlo[i] = (__bf16)(v - (float)hi);
}

// ---------------- kernel 3: h = x (vectorized copy) ----------------
__global__ void copy_x_kernel(const float4* __restrict__ x, float4* __restrict__ h,
                              long long n4) {
  long long i = (long long)blockIdx.x * blockDim.x + threadIdx.x;
  if (i < n4) h[i] = x[i];
}

// ---------------- kernel 4: edge scatter-add: h[dst,:] += x[src,:] ----------------
__global__ __launch_bounds__(256) void scatter_kernel(
    const float* __restrict__ x, const long long* __restrict__ ei,
    float* __restrict__ h, int E) {
  const int edge = blockIdx.x * (blockDim.x >> 5) + (threadIdx.x >> 5);
  if (edge >= E) return;
  const int lane = threadIdx.x & 31;
  const long long s = ei[edge];                 // edge_index[0, e]
  const long long d = ei[(long long)E + edge];  // edge_index[1, e]
  const float* __restrict__ xs = x + s * Fdim;
  float* __restrict__ hd = h + d * Fdim;
#pragma unroll
  for (int f = lane; f < Fdim; f += 32) {
    atomicAdd(&hd[f], xs[f]);
  }
}

// ---------------- kernel 5: pack h in place: fp32 -> (bf16_hi | bf16_lo<<16) ----------------
// hi+lo bf16 is exactly 4 bytes, so the packed form reuses the fp32 footprint.
__global__ void pack_hilo_kernel(float* __restrict__ h, long long n4) {
  long long i = (long long)blockIdx.x * blockDim.x + threadIdx.x;
  if (i >= n4) return;
  float4 f = ((const float4*)h)[i];
  uint4 w;
  w.x = pack_hilo(f.x);
  w.y = pack_hilo(f.y);
  w.z = pack_hilo(f.z);
  w.w = pack_hilo(f.w);
  ((uint4*)h)[i] = w;
}

// ---------------- kernel 6: z = h @ W^T + b via split-bf16 WMMA + BN stats ----------------
// Grid: (ceil(N/128), Hdim/64) blocks of 256 threads (8 waves).
// All 8 waves of a block share ONE 64-column group (blockIdx.y) so the per-chunk
// B working set (64 W-rows x 32 K x 2 planes = 8 KB) is WGP$-resident and reused
// 8x; each wave owns a distinct 16-row M-tile (blockIdx.x*8 + wave).
// A operand arrives pre-packed (hi|lo per dword): 1 v_perm_b32 per output dword.
// B operands arrive as pre-split bf16 planes: raw b128 loads, zero VALU.
union AVec { v16bf v; unsigned u[8]; };
union BVec { v16bf v; uint4 q[2]; };

__global__ __launch_bounds__(256) void gemm_stats_kernel(
    const unsigned* __restrict__ hp,                       // packed h, [N][Fdim]
    const __bf16* __restrict__ Whi, const __bf16* __restrict__ Wlo,
    const float* __restrict__ bias, float* __restrict__ z,
    float* __restrict__ colsum, float* __restrict__ colsumsq, int nMTiles) {
  const int lane  = threadIdx.x & 31;
  const int wave  = threadIdx.x >> 5;              // 0..7 -> M-tile within block
  const int mTile = blockIdx.x * 8 + wave;
  if (mTile >= nMTiles) return;                    // uniform per wave (tail block)
  const int mBase = mTile * 16;
  const int cGrp  = blockIdx.y;                    // column group of 64
  const int halfw = lane >> 4;                     // 0 or 1 (lane half)
  const int mn    = lane & 15;                     // row (A/C) or column (B/C)

  // C/D layout (ISA 7.12.2): element r holds M = r + 8*halfw, N = lane&15.
  v8f acc[4];
#pragma unroll
  for (int t = 0; t < 4; ++t) {
    const int col = (cGrp * 4 + t) * 16 + mn;
    const float bv = bias[col];
#pragma unroll
    for (int r = 0; r < 8; ++r) acc[t][r] = bv;
  }

  const unsigned* __restrict__ aRow = hp + (long long)(mBase + mn) * Fdim;

  for (int kc = 0; kc < KCHUNKS; ++kc) {
    const int k0 = kc * 32;

    // ---- A tile (16x32 bf16). Per-lane element e: K = (e&7) + 16*(e>>3) + 8*halfw.
    const unsigned* p = aRow + k0 + halfw * 8;
    uint4 p0 = *(const uint4*)(p + 0);
    uint4 p1 = *(const uint4*)(p + 4);
    uint4 p2 = *(const uint4*)(p + 16);
    uint4 p3 = *(const uint4*)(p + 20);
    unsigned u[16] = {p0.x, p0.y, p0.z, p0.w, p1.x, p1.y, p1.z, p1.w,
                      p2.x, p2.y, p2.z, p2.w, p3.x, p3.y, p3.z, p3.w};
    AVec ah, al;
#pragma unroll
    for (int j = 0; j < 8; ++j) {
      // out_hi = {p_odd[15:0],  p_even[15:0] }, out_lo = {p_odd[31:16], p_even[31:16]}
      ah.u[j] = __builtin_amdgcn_perm(u[2 * j + 1], u[2 * j], 0x05040100u);
      al.u[j] = __builtin_amdgcn_perm(u[2 * j + 1], u[2 * j], 0x07060302u);
    }

#pragma unroll
    for (int t = 0; t < 4; ++t) {
      // B tile (32x16 bf16): column n = lane&15 maps to W row; per-lane element e:
      // K = (e&7) + 16*(e>>3) + 8*halfw.  Two b128 loads per plane.
      const long long wrow = (cGrp * 4 + t) * 16 + mn;
      const long long off  = wrow * (long long)Fdim + k0 + halfw * 8;
      BVec bh, bl;
      bh.q[0] = *(const uint4*)(Whi + off);
      bh.q[1] = *(const uint4*)(Whi + off + 16);
      bl.q[0] = *(const uint4*)(Wlo + off);
      bl.q[1] = *(const uint4*)(Wlo + off + 16);
      // Split-precision product: A*B ~= Ahi*Bhi + Ahi*Blo + Alo*Bhi (fp32 accum).
      acc[t] = __builtin_amdgcn_wmma_f32_16x16x32_bf16(
          false, ah.v, false, bh.v, (short)0, acc[t], false, false);
      acc[t] = __builtin_amdgcn_wmma_f32_16x16x32_bf16(
          false, ah.v, false, bl.v, (short)0, acc[t], false, false);
      acc[t] = __builtin_amdgcn_wmma_f32_16x16x32_bf16(
          false, al.v, false, bh.v, (short)0, acc[t], false, false);
    }
  }

  // ---- store z and accumulate per-column BN partial sums ----
#pragma unroll
  for (int t = 0; t < 4; ++t) {
    const int col = (cGrp * 4 + t) * 16 + mn;
    float s = 0.0f, ss = 0.0f;
#pragma unroll
    for (int r = 0; r < 8; ++r) {
      const float v = acc[t][r];
      z[(long long)(mBase + halfw * 8 + r) * Hdim + col] = v;
      s  += v;
      ss += v * v;
    }
    // lanes L and L+16 hold the same column: reduce across halves, one atomic.
    s  += __shfl_xor(s, 16, 32);
    ss += __shfl_xor(ss, 16, 32);
    if (halfw == 0) {
      atomicAdd(&colsum[col], s);
      atomicAdd(&colsumsq[col], ss);
    }
  }
}

// ---------------- kernel 7: BatchNorm finalize (in place over z/out) ----------------
__global__ void bn_kernel(const float* __restrict__ z,
                          const float* __restrict__ colsum,
                          const float* __restrict__ colsumsq,
                          const float* __restrict__ gamma,
                          const float* __restrict__ beta,
                          float* __restrict__ out, int N) {
  const long long i = ((long long)blockIdx.x * blockDim.x + threadIdx.x) * 4;
  if (i >= (long long)N * Hdim) return;
  const int col = (int)(i & (Hdim - 1));    // Hdim power of two; i multiple of 4
  const float invN = 1.0f / (float)N;
  float4 zv = *(const float4*)(z + i);
  float zz[4] = {zv.x, zv.y, zv.z, zv.w};
  float o[4];
#pragma unroll
  for (int c = 0; c < 4; ++c) {
    const int cc = col + c;
    const float mean = colsum[cc] * invN;
    const float var  = colsumsq[cc] * invN - mean * mean;
    o[c] = gamma[cc] * (zz[c] - mean) * rsqrtf(var + BN_EPS) + beta[cc];
  }
  *(float4*)(out + i) = make_float4(o[0], o[1], o[2], o[3]);
}

extern "C" void kernel_launch(void* const* d_in, const int* in_sizes, int n_in,
                              void* d_out, int out_size, void* d_ws, size_t ws_size,
                              hipStream_t stream) {
  (void)n_in; (void)out_size; (void)ws_size;
  const float*     x     = (const float*)d_in[0];
  const long long* ei    = (const long long*)d_in[1];   // int64 edge_index [2, E]
  /* d_in[2] adj_norm_sp is unused by the forward pass */
  const float*     W     = (const float*)d_in[3];
  const float*     b     = (const float*)d_in[4];
  const float*     gamma = (const float*)d_in[5];
  const float*     beta  = (const float*)d_in[6];
  float*           out   = (float*)d_out;

  const int N = in_sizes[0] / Fdim;   // 50000 (multiple of 16)
  const int E = in_sizes[1] / 2;      // 160000

  // Workspace layout:
  //   h[N*Fdim] (fp32, later packed hi|lo in place) | colsum[Hdim] | colsumsq[Hdim]
  //   | Whi[Hdim*Fdim] bf16 | Wlo[Hdim*Fdim] bf16
  float*  h        = (float*)d_ws;
  float*  colsum   = h + (long long)N * Fdim;
  float*  colsumsq = colsum + Hdim;
  __bf16* Whi      = (__bf16*)(colsumsq + Hdim);
  __bf16* Wlo      = Whi + (long long)Hdim * Fdim;

  // 1. zero BN accumulators
  zero_stats_kernel<<<(2 * Hdim + 255) / 256, 256, 0, stream>>>(colsum, 2 * Hdim);

  // 2. split W into bf16 hi/lo planes (Hdim*Fdim = 256K elements)
  wsplit_kernel<<<(Hdim * Fdim + 255) / 256, 256, 0, stream>>>(W, Whi, Wlo,
                                                               Hdim * Fdim);

  // 3. h = x
  const long long n4 = (long long)N * Fdim / 4;
  copy_x_kernel<<<(int)((n4 + 255) / 256), 256, 0, stream>>>(
      (const float4*)x, (float4*)h, n4);

  // 4. scatter-add (wave per edge, 8 edges per block)
  scatter_kernel<<<(E + 7) / 8, 256, 0, stream>>>(x, ei, h, E);

  // 5. pack h in place to (bf16_hi | bf16_lo<<16)
  pack_hilo_kernel<<<(int)((n4 + 255) / 256), 256, 0, stream>>>(h, n4);

  // 6. WMMA GEMM + bias + BN partial stats (z written to d_out)
  //    grid: x = M-tile groups of 8 waves, y = 64-column groups
  const int nMTiles = N / 16;
  dim3 ggrid((nMTiles + 7) / 8, Hdim / 64);
  gemm_stats_kernel<<<ggrid, 256, 0, stream>>>((const unsigned*)h, Whi, Wlo, b,
                                               out, colsum, colsumsq, nMTiles);

  // 7. BatchNorm finalize in place
  const long long tot4 = (long long)N * Hdim / 4;
  bn_kernel<<<(int)((tot4 + 255) / 256), 256, 0, stream>>>(
      out, colsum, colsumsq, gamma, beta, out, N);
}
